// Memory_22548578304755
// MI455X (gfx1250) — compile-verified, matrix operand order checked by reading, not run on hardware
//
#include <hip/hip_runtime.h>
#include <hip/hip_bf16.h>
#include <math.h>

typedef __attribute__((ext_vector_type(16))) __bf16 v16bf;
typedef __attribute__((ext_vector_type(8)))  float  v8f;
typedef __attribute__((ext_vector_type(4)))  float  v4f;

#define BLK           256
#define WAVES         8
#define COLS_PER_WAVE 32
#define COLS_PER_WG   256
#define BATCH         64
#define TEMP_INV      (1.0f / 0.07f)
#define NEG_INF       (-__builtin_inff())

__device__ __forceinline__ float rmax16(float v) {
    #pragma unroll
    for (int off = 1; off < 16; off <<= 1) v = fmaxf(v, __shfl_xor(v, off, 16));
    return v;
}
__device__ __forceinline__ float radd16(float v) {
    #pragma unroll
    for (int off = 1; off < 16; off <<= 1) v += __shfl_xor(v, off, 16);
    return v;
}
// online-softmax combine: (M,S) <- (M,S) merge (gm,gs)
__device__ __forceinline__ void lse_combine(float& M, float& S, float gm, float gs) {
    float Mn = fmaxf(M, gm);
    if (Mn > NEG_INF) {
        float Sa = (S  > 0.f) ? S  * __expf(M  - Mn) : 0.f;
        float Sb = (gs > 0.f) ? gs * __expf(gm - Mn) : 0.f;
        M = Mn; S = Sa + Sb;
    }
}

// -------- Kernel 0: one-shot fp32 -> bf16 conversion of inputs (stays L2-resident) --------
__global__ void __launch_bounds__(BLK)
cvt_inputs_kernel(const float* __restrict__ in, __bf16* __restrict__ out, int n) {
    int i = blockIdx.x * BLK + threadIdx.x;
    if (i < n) out[i] = (__bf16)in[i];
}

// -------- Kernel 1: bf16 WMMA GEMM + masked per-row (max, sumexp) partials --------
// A (inputs_bf16, 64xD) read from global via 32-bit offsets (L2-resident, 256 KB).
// B (features, NxD fp32) streamed once from HBM with non-temporal loads, cvt to bf16.
__global__ void __launch_bounds__(BLK)
nce_partial_kernel(const __bf16* __restrict__ inputs_bf,   // [64, D] bf16
                   const float*  __restrict__ features,    // [N, D] fp32
                   const int*    __restrict__ camids,      // [N]
                   const int*    __restrict__ camids_batch,// [64]
                   float*        __restrict__ partials,    // [nChunks, 64, 2]
                   int N, int D)
{
    __shared__ float redM[WAVES][BATCH];
    __shared__ float redS[WAVES][BATCH];
    __shared__ int   cb_s[BATCH];

    const int tid  = threadIdx.x;
    const int wave = tid >> 5;
    const int lane = tid & 31;
    const int l16  = lane & 15;
    const int half = lane >> 4;       // K-half: lanes 0-15 -> K lo16, lanes 16-31 -> K hi16

    if (tid < BATCH) cb_s[tid] = camids_batch[tid];

    const int cbase = blockIdx.x * COLS_PER_WG + wave * COLS_PER_WAVE;

    // column indices (clamped; mask excludes clamped columns via camn = -1)
    int ncol[2], camn[2];
    #pragma unroll
    for (int g = 0; g < 2; ++g) {
        int n   = cbase + g * 16 + l16;
        camn[g] = (n < N) ? camids[n] : -1;
        ncol[g] = (n < N) ? n : (N - 1);
    }

    // 32-bit element offsets from uniform bases -> GVS-form global_load_b128
    const v4f*   fbase = reinterpret_cast<const v4f*>(features);     // float4 units
    const v16bf* abase = reinterpret_cast<const v16bf*>(inputs_bf);  // 16-bf16 units
    int f0 = (ncol[0] * D + half * 16) >> 2;     // float4 index (fits in i32: < 52M)
    int f1 = (ncol[1] * D + half * 16) >> 2;
    int aoff[4];
    #pragma unroll
    for (int mt = 0; mt < 4; ++mt)
        aoff[mt] = ((mt * 16 + l16) * D + half * 16) >> 4;  // v16bf index

    v8f acc[4][2];
    #pragma unroll
    for (int mt = 0; mt < 4; ++mt)
        #pragma unroll
        for (int g = 0; g < 2; ++g)
            #pragma unroll
            for (int e = 0; e < 8; ++e) acc[mt][g][e] = 0.f;

    #pragma unroll 2
    for (int k = 0; k < D; k += 32) {
        // Phase 1: issue ALL loads with independent dest regs (one big clause in flight)
        v4f r[8];
        #pragma unroll
        for (int j = 0; j < 4; ++j) r[j]     = __builtin_nontemporal_load(fbase + f0 + j);
        #pragma unroll
        for (int j = 0; j < 4; ++j) r[4 + j] = __builtin_nontemporal_load(fbase + f1 + j);
        v16bf a[4];
        #pragma unroll
        for (int mt = 0; mt < 4; ++mt) a[mt] = abase[aoff[mt]];
        __builtin_prefetch(reinterpret_cast<const float*>(fbase + f0) + 256, 0, 1);
        __builtin_prefetch(reinterpret_cast<const float*>(fbase + f1) + 256, 0, 1);
        f0 += 8; f1 += 8;
        #pragma unroll
        for (int mt = 0; mt < 4; ++mt) aoff[mt] += 2;

        // Phase 2: fp32 -> bf16 conversion of the streamed B fragments
        v16bf bf0, bf1;
        #pragma unroll
        for (int j = 0; j < 4; ++j)
            #pragma unroll
            for (int e = 0; e < 4; ++e) {
                bf0[j * 4 + e] = (__bf16)r[j][e];
                bf1[j * 4 + e] = (__bf16)r[4 + j][e];
            }

        // Phase 3: 8 WMMAs (A fragments reused across both N-groups)
        #pragma unroll
        for (int mt = 0; mt < 4; ++mt) {
            acc[mt][0] = __builtin_amdgcn_wmma_f32_16x16x32_bf16(
                false, a[mt], false, bf0, (short)0, acc[mt][0], false, false);
            acc[mt][1] = __builtin_amdgcn_wmma_f32_16x16x32_bf16(
                false, a[mt], false, bf1, (short)0, acc[mt][1], false, false);
        }
    }

    // C/D layout: lane = N (0..15 per half), VGPR slot s + 8*half = M within tile
    #pragma unroll
    for (int mt = 0; mt < 4; ++mt) {
        #pragma unroll
        for (int s = 0; s < 8; ++s) {
            const int m  = mt * 16 + s + half * 8;
            const int cb = cb_s[m];
            float M = NEG_INF, S = 0.f;
            #pragma unroll
            for (int g = 0; g < 2; ++g) {
                float v  = (camn[g] == cb) ? acc[mt][g][s] * TEMP_INV : NEG_INF;
                float gm = rmax16(v);
                float p  = (v > NEG_INF) ? __expf(v - gm) : 0.f;
                float gs = radd16(p);
                lse_combine(M, S, gm, gs);
            }
            if (l16 == 0) { redM[wave][m] = M; redS[wave][m] = S; }
        }
    }
    __syncthreads();

    // merge 8 waves -> one (max, sumexp) per batch row for this chunk
    if (tid < BATCH) {
        float M = NEG_INF, S = 0.f;
        #pragma unroll
        for (int w = 0; w < WAVES; ++w) lse_combine(M, S, redM[w][tid], redS[w][tid]);
        float2* out = reinterpret_cast<float2*>(partials);
        out[(size_t)blockIdx.x * BATCH + tid] = make_float2(M, S);
    }
}

// -------- Kernel 2: reduce partials, target dots, final scalar --------
__global__ void __launch_bounds__(BLK)
nce_finalize_kernel(const float*  __restrict__ inputs,
                    const float*  __restrict__ features,
                    const int*    __restrict__ indices,
                    const float2* __restrict__ partials,
                    float*        __restrict__ out,
                    int nChunks, int D)
{
    __shared__ float lse[BATCH];
    __shared__ float tdot[BATCH];
    __shared__ float red[BLK];
    const int tid = threadIdx.x;

    if (tid < BATCH) {
        float M = NEG_INF, S = 0.f;
        for (int c = 0; c < nChunks; ++c) {
            float2 p = partials[(size_t)c * BATCH + tid];
            lse_combine(M, S, p.x, p.y);
        }
        lse[tid] = M + __logf(fmaxf(S, 1e-37f));
    }
    __syncthreads();

    for (int b = 0; b < BATCH; ++b) {
        const int ib = indices[b];
        float p = 0.f;
        for (int k = tid; k < D; k += BLK)
            p += inputs[(size_t)b * D + k] * features[(size_t)ib * D + k];
        red[tid] = p;
        __syncthreads();
        #pragma unroll
        for (int off = BLK / 2; off > 0; off >>= 1) {
            if (tid < off) red[tid] += red[tid + off];
            __syncthreads();
        }
        if (tid == 0) tdot[b] = red[0] * TEMP_INV;
        __syncthreads();
    }

    if (tid == 0) {
        float loss = 0.f;
        for (int b = 0; b < BATCH; ++b) loss += lse[b] - tdot[b];
        out[0] = loss / (float)BATCH;
    }
}

extern "C" void kernel_launch(void* const* d_in, const int* in_sizes, int n_in,
                              void* d_out, int out_size, void* d_ws, size_t ws_size,
                              hipStream_t stream) {
    const float* inputs       = (const float*)d_in[0];
    const float* features     = (const float*)d_in[1];
    const int*   indices      = (const int*)d_in[2];
    const int*   camids       = (const int*)d_in[3];
    const int*   camids_batch = (const int*)d_in[4];

    const int B = BATCH;
    const int D = in_sizes[0] / B;     // 2048
    const int N = in_sizes[3];         // 100000
    const int nChunks = (N + COLS_PER_WG - 1) / COLS_PER_WG;

    // ws layout: [0, 64*D) bf16 inputs ; then partials (float2 per chunk-row)
    __bf16* inputs_bf = (__bf16*)d_ws;
    float*  partials  = (float*)((char*)d_ws + (size_t)B * D * sizeof(__bf16));

    const int nElem = B * D;
    cvt_inputs_kernel<<<(nElem + BLK - 1) / BLK, BLK, 0, stream>>>(inputs, inputs_bf, nElem);
    nce_partial_kernel<<<nChunks, BLK, 0, stream>>>(
        inputs_bf, features, camids, camids_batch, partials, N, D);
    nce_finalize_kernel<<<1, BLK, 0, stream>>>(
        inputs, features, indices, (const float2*)partials, (float*)d_out, nChunks, D);
}
